// HPB_54494545052430
// MI455X (gfx1250) — matrix operand order repaired and checked by reading, not compiled, not run on hardware
//
#include <hip/hip_runtime.h>
#include <hip/hip_bf16.h>
#include <stdint.h>

typedef __attribute__((ext_vector_type(16))) _Float16 v16h;
typedef __attribute__((ext_vector_type(8)))  _Float16 v8h;
typedef __attribute__((ext_vector_type(8)))  float    v8f;
typedef __attribute__((ext_vector_type(4)))  float    v4f;

union V16U { v16h v; v8h h[2]; };

#define DEV __device__ __forceinline__

constexpr int C_   = 192;
constexpr int NH_  = 8;
constexpr int HC_  = 64;
constexpr int HID_ = 512;
constexpr int P_   = 16;
constexpr int B_   = 2;
constexpr int HW_  = 128 * 128;       // 16384
constexpr int NTOT = B_ * HW_;        // 32768
constexpr int BH_  = B_ * NH_;        // 16
constexpr float EPS_ = 1e-5f;

// ---------------------------------------------------------------- WMMA core
DEV v8f wmma32(v16h a, v16h b, v8f c) {
  return __builtin_amdgcn_wmma_f32_16x16x32_f16(false, a, false, b, (short)0, c,
                                                false, false);
}

// A fragment (16x32 f16) from an f16 weight matrix, row-major, ld = K.
// lane<16: row m0+lane, K=kk+0..15; lane>=16: K=kk+16..31.
DEV v16h load_a_h(const _Float16* __restrict__ Wp, int ld, int m0, int kk, int lane) {
  int row = m0 + (lane & 15);
  int kb  = kk + ((lane & 16) ? 16 : 0);
  const _Float16* p = Wp + (size_t)row * ld + kb;
  V16U u;
  u.h[0] = *(const v8h*)p;
  u.h[1] = *(const v8h*)(p + 8);
  return u.v;
}

// B fragment (32x16 f16) from an f16 matrix stored (N, K) with K contiguous.
// lane<16: col n0+lane, K=kk+0..15; lane>=16: K=kk+16..31. One 32B load.
DEV v16h load_b_x(const _Float16* __restrict__ X, int ld, int n0, int kk, int lane) {
  int col = n0 + (lane & 15);
  int kb  = kk + ((lane & 16) ? 16 : 0);
  return *(const v16h*)(X + (size_t)col * ld + kb);
}

DEV float gelu_exact(float v) { return 0.5f * v * (1.0f + erff(v * 0.70710678118654752f)); }

// ---------------------------------------------------------------- generic GEMM
// One wave computes a 16(M) x 64(N) strip: A fragment loaded once per k-step,
// 4 back-to-back WMMAs against 4 B fragments.
enum { MODE_HEADSPLIT = 0, MODE_CAT = 1, MODE_MLPF = 2, MODE_NCHW = 3 };

template <int MODE>
__global__ void __launch_bounds__(32)
gemm16(const _Float16* __restrict__ A, const _Float16* __restrict__ Bx,
       const float* __restrict__ bias, const float* __restrict__ resid,
       float* __restrict__ outF, _Float16* __restrict__ outH, int K) {
  const int lane = threadIdx.x;
  const int n0 = blockIdx.x * 64;
  const int m0 = blockIdx.y * 16;
  v8f acc[4] = {};
  for (int kk = 0; kk < K; kk += 32) {
    v16h a = load_a_h(A, K, m0, kk, lane);
#pragma unroll
    for (int t = 0; t < 4; t++) {
      v16h b = load_b_x(Bx, K, n0 + t * 16, kk, lane);
      acc[t] = wmma32(a, b, acc[t]);
    }
  }
  const int mrow = m0 + ((lane & 16) ? 8 : 0);

  float badd[8];
  if constexpr (MODE != MODE_HEADSPLIT) {
#pragma unroll
    for (int r = 0; r < 8; r++) badd[r] = bias[mrow + r];
  }

#pragma unroll
  for (int t = 0; t < 4; t++) {
    const int n = n0 + t * 16 + (lane & 15);
    const int bidx = n >> 14;          // batch (HW_ = 2^14)
    const int hw = n & (HW_ - 1);

    if constexpr (MODE == MODE_HEADSPLIT) {
      const int head = m0 >> 6;
      const int c0 = mrow & 63;
      _Float16* dst = outH + (((size_t)(bidx * NH_ + head) * HW_ + hw) * HC_ + c0);
      v8h hv;
#pragma unroll
      for (int r = 0; r < 8; r++) hv[r] = (_Float16)acc[t][r];
      *(v8h*)dst = hv;
    } else if constexpr (MODE == MODE_CAT) {
      _Float16* dst = outH + ((size_t)n * (2 * C_) + C_ + mrow);
      v8h hv;
#pragma unroll
      for (int r = 0; r < 8; r++) hv[r] = (_Float16)(acc[t][r] + badd[r]);
      *(v8h*)dst = hv;
    } else if constexpr (MODE == MODE_MLPF) {
      float* dst = outF + ((size_t)n * C_ + mrow);
      v4f lo, hi;
#pragma unroll
      for (int r = 0; r < 4; r++) {
        lo[r] = acc[t][r]     + badd[r]     + resid[((size_t)bidx * C_ + mrow + r) * HW_ + hw];
        hi[r] = acc[t][4 + r] + badd[4 + r] + resid[((size_t)bidx * C_ + mrow + 4 + r) * HW_ + hw];
      }
      *(v4f*)dst = lo;
      *(v4f*)(dst + 4) = hi;
    } else {  // MODE_NCHW
#pragma unroll
      for (int r = 0; r < 8; r++) {
        outF[((size_t)bidx * C_ + mrow + r) * HW_ + hw] = acc[t][r] + badd[r];
      }
    }
  }
}

// V projection only at the 256 gathered pixels per head -> vgT (bh, 64, 256)
__global__ void __launch_bounds__(32)
gemm_vg(const _Float16* __restrict__ wv, const _Float16* __restrict__ xn,
        const int* __restrict__ gidx, _Float16* __restrict__ vgT) {
  const int lane = threadIdx.x;
  const int bh = blockIdx.z;
  const int n0 = blockIdx.x * 16;   // key tile
  const int m0 = blockIdx.y * 16;   // channel tile within head
  const int head = bh & (NH_ - 1);
  const int key = n0 + (lane & 15);
  const int gn = gidx[bh * 256 + key];
  const _Float16* brow = xn + (size_t)gn * C_;
  v8f acc = {};
  for (int kk = 0; kk < C_; kk += 32) {
    v16h a = load_a_h(wv + (size_t)head * HC_ * C_, C_, m0, kk, lane);
    int kb = kk + ((lane & 16) ? 16 : 0);
    v16h b = *(const v16h*)(brow + kb);
    acc = wmma32(a, b, acc);
  }
  const int mrow = m0 + ((lane & 16) ? 8 : 0);
#pragma unroll
  for (int r = 0; r < 8; r++)
    vgT[((size_t)bh * HC_ + mrow + r) * 256 + key] = (_Float16)acc[r];
}

// ---------------------------------------------------------------- attention
// One wave handles 16 query pixels of one head: S = q@kg^T (16x256) -> softmax
// -> O = P@vg (16x64), 1/rowsum folded into the store.
__global__ void __launch_bounds__(32)
attn_kernel(const _Float16* __restrict__ q16, const _Float16* __restrict__ kg,
            const _Float16* __restrict__ vgT, _Float16* __restrict__ att) {
  __shared__ float S[16 * 256];
  __shared__ _Float16 Ph[16 * 256];
  __shared__ float rsum[16];
  const int lane = threadIdx.x;
  const int bh = blockIdx.y;
  const int p0 = blockIdx.x * 16;
  const int m = lane & 15;
  const int m_base = (lane & 16) ? 8 : 0;
  const int ko = (lane & 16) ? 16 : 0;

  // q A-fragments for K = 0..31 and 32..63
  const _Float16* qp = q16 + ((size_t)bh * HW_ + p0 + m) * HC_;
  V16U aq0, aq1;
  aq0.h[0] = *(const v8h*)(qp + 0 + ko);
  aq0.h[1] = *(const v8h*)(qp + 8 + ko);
  aq1.h[0] = *(const v8h*)(qp + 32 + ko);
  aq1.h[1] = *(const v8h*)(qp + 40 + ko);

  // S tiles
  for (int nt = 0; nt < 16; nt++) {
    const _Float16* krow = kg + ((size_t)bh * 256 + nt * 16 + m) * HC_;
    int kbo = (lane & 16) ? 16 : 0;
    v8f acc = {};
    acc = wmma32(aq0.v, *(const v16h*)(krow + kbo), acc);
    acc = wmma32(aq1.v, *(const v16h*)(krow + 32 + kbo), acc);
    int ccol = nt * 16 + m;
#pragma unroll
    for (int r = 0; r < 8; r++) S[(m_base + r) * 256 + ccol] = acc[r];
  }
  __syncthreads();

  // softmax over 256 keys: lane pair (L, L^16) handles row m, 128 cols each
  const int j0 = (lane & 16) ? 128 : 0;
  float mx = -3.0e38f;
  for (int j = 0; j < 128; j++) mx = fmaxf(mx, S[m * 256 + j0 + j]);
  mx = fmaxf(mx, __shfl_xor(mx, 16, 32));
  float ssum = 0.f;
  for (int j = 0; j < 128; j++) {
    float e = __expf(S[m * 256 + j0 + j] - mx);
    Ph[m * 256 + j0 + j] = (_Float16)e;
    ssum += e;
  }
  ssum += __shfl_xor(ssum, 16, 32);
  if (!(lane & 16)) rsum[m] = ssum;
  __syncthreads();

  // O = P @ vg  (M=16 queries, N=64 channels, K=256 keys)
  v8f oacc[4] = {};
  for (int kk = 0; kk < 256; kk += 32) {
    V16U a;
    const _Float16* pp = Ph + m * 256 + kk + ko;
    a.h[0] = *(const v8h*)pp;
    a.h[1] = *(const v8h*)(pp + 8);
    int kbo = (lane & 16) ? 16 : 0;
#pragma unroll
    for (int ct = 0; ct < 4; ct++) {
      const _Float16* vp = vgT + ((size_t)bh * HC_ + ct * 16 + m) * 256 + kk + kbo;
      oacc[ct] = wmma32(a.v, *(const v16h*)vp, oacc[ct]);
    }
  }
  float inv[8];
#pragma unroll
  for (int r = 0; r < 8; r++) inv[r] = 1.f / rsum[m_base + r];
  const int bb = bh >> 3, head = bh & 7;
#pragma unroll
  for (int ct = 0; ct < 4; ct++)
#pragma unroll
    for (int r = 0; r < 8; r++) {
      int pix = p0 + m_base + r;
      att[((size_t)bb * HW_ + pix) * HID_ + head * HC_ + ct * 16 + m] =
          (_Float16)(oacc[ct][r] * inv[r]);
    }
}

// ---------------------------------------------------------------- small kernels
__global__ void __launch_bounds__(256)
cvt_f16_kernel(const float* __restrict__ src, _Float16* __restrict__ dst, int count) {
  int i = blockIdx.x * 256 + threadIdx.x;
  if (i < count) dst[i] = (_Float16)src[i];
}

__global__ void __launch_bounds__(256)
ln1_kernel(const float* __restrict__ x, const float* __restrict__ g,
           const float* __restrict__ b, _Float16* __restrict__ xn) {
  int n = blockIdx.x * 256 + threadIdx.x;
  if (n >= NTOT) return;
  int bb = n >> 14, hw = n & (HW_ - 1);
  const float* xp = x + (size_t)bb * C_ * HW_ + hw;
  float s = 0.f, s2 = 0.f;
  for (int c = 0; c < C_; c++) { float v = xp[(size_t)c * HW_]; s += v; s2 += v * v; }
  float mean = s * (1.f / C_);
  float inv = rsqrtf(s2 * (1.f / C_) - mean * mean + EPS_);
  _Float16* o = xn + (size_t)n * C_;
  for (int c = 0; c < C_; c++) {
    float v = xp[(size_t)c * HW_];
    o[c] = (_Float16)((v - mean) * inv * g[c] + b[c]);
  }
}

__global__ void __launch_bounds__(256)
ln2_kernel(const float* __restrict__ f, const float* __restrict__ g,
           const float* __restrict__ b, _Float16* __restrict__ o16) {
  int n = blockIdx.x * 256 + threadIdx.x;
  if (n >= NTOT) return;
  const float* fp = f + (size_t)n * C_;
  float s = 0.f, s2 = 0.f;
  for (int c = 0; c < C_; c++) { float v = fp[c]; s += v; s2 += v * v; }
  float mean = s * (1.f / C_);
  float inv = rsqrtf(s2 * (1.f / C_) - mean * mean + EPS_);
  _Float16* o = o16 + (size_t)n * C_;
  for (int c = 0; c < C_; c++) o[c] = (_Float16)((fp[c] - mean) * inv * g[c] + b[c]);
}

__global__ void __launch_bounds__(256)
l2norm_kernel(_Float16* __restrict__ t) {
  int n = blockIdx.x * 256 + threadIdx.x;
  if (n >= BH_ * HW_) return;
  _Float16* p = t + (size_t)n * HC_;
  float s = 0.f;
  for (int i = 0; i < HC_; i++) { float v = (float)p[i]; s += v * v; }
  float inv = 1.f / fmaxf(sqrtf(s), 1e-12f);
  for (int i = 0; i < HC_; i++) p[i] = (_Float16)((float)p[i] * inv);
}

__global__ void __launch_bounds__(256)
qs_kernel(const _Float16* __restrict__ q, float* __restrict__ qs) {
  int bh = blockIdx.x, t = threadIdx.x;
  int c = t & 63, chunk = t >> 6;
  float s = 0.f;
  for (int p = chunk; p < HW_; p += 4)
    s += (float)q[((size_t)bh * HW_ + p) * HC_ + c];
  __shared__ float red[256];
  red[t] = s;
  __syncthreads();
  if (t < 64) qs[bh * HC_ + t] = red[t] + red[t + 64] + red[t + 128] + red[t + 192];
}

__global__ void __launch_bounds__(64)
scores_kernel(const _Float16* __restrict__ k, const float* __restrict__ qs,
              float* __restrict__ hs, float* __restrict__ ws) {
  int line = blockIdx.x, bh = blockIdx.y, dir = blockIdx.z, c = threadIdx.x;
  float s = 0.f;
  for (int i = 0; i < 128; i++) {
    int pix = (dir == 0) ? (line * 128 + i) : (i * 128 + line);
    s += (float)k[((size_t)bh * HW_ + pix) * HC_ + c];
  }
  s *= qs[bh * HC_ + c];
  __shared__ float red[64];
  red[c] = s;
  __syncthreads();
  for (int st = 32; st > 0; st >>= 1) { if (c < st) red[c] += red[c + st]; __syncthreads(); }
  if (c == 0) (dir == 0 ? hs : ws)[bh * 128 + line] = red[0];
}

__global__ void topk_kernel(const float* __restrict__ hs, const float* __restrict__ ws,
                            int* __restrict__ hidx, int* __restrict__ widx) {
  int bh = blockIdx.x;
  int which = threadIdx.x;
  if (which >= 2) return;
  const float* sc = which ? (ws + bh * 128) : (hs + bh * 128);
  int* oi = which ? (widx + bh * P_) : (hidx + bh * P_);
  unsigned long long used0 = 0, used1 = 0;
  for (int t = 0; t < P_; t++) {
    float best = -3.4e38f; int bi = 0;
    for (int i = 0; i < 128; i++) {
      bool u = (i < 64) ? ((used0 >> i) & 1ULL) : ((used1 >> (i - 64)) & 1ULL);
      if (u) continue;
      float v = sc[i];
      if (v > best) { best = v; bi = i; }
    }
    if (bi < 64) used0 |= 1ULL << bi; else used1 |= 1ULL << (bi - 64);
    oi[t] = bi;
  }
}

__global__ void __launch_bounds__(256)
gather_kernel(const _Float16* __restrict__ k16, const int* __restrict__ hidx,
              const int* __restrict__ widx, _Float16* __restrict__ kg,
              int* __restrict__ gidx) {
  int bh = blockIdx.x, key = threadIdx.x;
  int hi = hidx[bh * P_ + (key >> 4)];
  int wi = widx[bh * P_ + (key & 15)];
  int pix = hi * 128 + wi;
  gidx[bh * 256 + key] = (bh >> 3) * HW_ + pix;
  const v8h* src = (const v8h*)(k16 + ((size_t)bh * HW_ + pix) * HC_);
  v8h* dst = (v8h*)(kg + ((size_t)bh * 256 + key) * HC_);
#pragma unroll
  for (int i = 0; i < 8; i++) dst[i] = src[i];
}

template <int OUT_CAT>
__global__ void __launch_bounds__(256)
dwconv_kernel(const float* __restrict__ x, const float* __restrict__ w,
              const float* __restrict__ bias, _Float16* __restrict__ outH,
              float* __restrict__ outF) {
  int idx = blockIdx.x * 256 + threadIdx.x;
  if (idx >= B_ * C_ * HW_) return;
  int hw = idx & (HW_ - 1);
  int bc = idx >> 14;
  int c = bc % C_;
  int b = bc / C_;
  int h = hw >> 7, wc = hw & 127;
  const float* xp = x + (size_t)bc * HW_;
  const float* wp = w + c * 9;
  float s = bias[c];
#pragma unroll
  for (int dh = -1; dh <= 1; dh++) {
    int hh = h + dh;
    if ((unsigned)hh >= 128u) continue;
#pragma unroll
    for (int dw = -1; dw <= 1; dw++) {
      int ww = wc + dw;
      if ((unsigned)ww >= 128u) continue;
      s += wp[(dh + 1) * 3 + (dw + 1)] * xp[hh * 128 + ww];
    }
  }
  if constexpr (OUT_CAT)
    outH[((size_t)b * HW_ + hw) * (2 * C_) + c] = (_Float16)s;
  else
    outF[idx] = s;
}

// MODE 0: gelu(inorm) in place; 1: gelu(inorm)+resid -> f16 (n,c); 2: inorm -> outF
template <int MODE>
__global__ void __launch_bounds__(256)
inorm_kernel(float* __restrict__ data, const float* __restrict__ resid,
             _Float16* __restrict__ outH, float* __restrict__ outF) {
  const int plane = blockIdx.x;  // b*C_ + c
  const int t = threadIdx.x;
  float* p = data + (size_t)plane * HW_;
  float s = 0.f, s2 = 0.f;
  for (int i = t; i < HW_; i += 256) { float v = p[i]; s += v; s2 += v * v; }
  __shared__ float rs[256], rq[256];
  rs[t] = s; rq[t] = s2;
  __syncthreads();
  for (int st = 128; st > 0; st >>= 1) {
    if (t < st) { rs[t] += rs[t + st]; rq[t] += rq[t + st]; }
    __syncthreads();
  }
  const float mean = rs[0] * (1.f / HW_);
  const float var = rq[0] * (1.f / HW_) - mean * mean;
  const float inv = rsqrtf(var + EPS_);
  const int c = plane % C_;
  const int b = plane / C_;
  for (int i = t; i < HW_; i += 256) {
    float v = (p[i] - mean) * inv;
    if constexpr (MODE == 0) {
      p[i] = gelu_exact(v);
    } else if constexpr (MODE == 1) {
      outH[((size_t)b * HW_ + i) * C_ + c] = (_Float16)(gelu_exact(v) + resid[(size_t)plane * HW_ + i]);
    } else {
      outF[(size_t)plane * HW_ + i] = v;
    }
  }
}

// ---------------------------------------------------------------- launch
extern "C" void kernel_launch(void* const* d_in, const int* in_sizes, int n_in,
                              void* d_out, int out_size, void* d_ws, size_t ws_size,
                              hipStream_t stream) {
  (void)in_sizes; (void)n_in; (void)out_size; (void)ws_size;
  const float* x    = (const float*)d_in[0];
  const float* ln1g = (const float*)d_in[1];
  const float* ln1b = (const float*)d_in[2];
  const float* wq   = (const float*)d_in[3];
  const float* wk   = (const float*)d_in[4];
  const float* wv   = (const float*)d_in[5];
  const float* wo   = (const float*)d_in[6];
  const float* bo   = (const float*)d_in[7];
  const float* dww  = (const float*)d_in[8];
  const float* dwb  = (const float*)d_in[9];
  const float* mlpw = (const float*)d_in[10];
  const float* mlpb = (const float*)d_in[11];
  const float* ln2g = (const float*)d_in[12];
  const float* ln2b = (const float*)d_in[13];
  const float* c1w  = (const float*)d_in[14];
  const float* c1b  = (const float*)d_in[15];
  const float* resw = (const float*)d_in[16];
  const float* resb = (const float*)d_in[17];
  const float* c2w  = (const float*)d_in[18];
  const float* c2b  = (const float*)d_in[19];
  float* out = (float*)d_out;

  uint8_t* wsb = (uint8_t*)d_ws;
  size_t off = 0;
  auto take = [&](size_t bytes) -> uint8_t* {
    uint8_t* p = wsb + off;
    off = (off + bytes + 255) & ~(size_t)255;
    return p;
  };
  _Float16* xn_t = (_Float16*)take((size_t)NTOT * C_ * 2);       // 12.6 MB
  uint8_t* R1 = take((size_t)BH_ * HW_ * HC_ * 2);               // q16  / buf0 (33.5 MB)
  uint8_t* R2 = take((size_t)BH_ * HW_ * HC_ * 2);               // k16  / buf1
  uint8_t* R3 = take((size_t)B_ * HW_ * HID_ * 2);               // att16 / f_t
  uint8_t* R4 = take((size_t)NTOT * 2 * C_ * 2);                 // cat16 / t16 (25.2 MB)
  float* qs  = (float*)take(BH_ * HC_ * 4);
  float* hs  = (float*)take(BH_ * 128 * 4);
  float* wsS = (float*)take(BH_ * 128 * 4);
  int* hidx  = (int*)take(BH_ * P_ * 4);
  int* widx  = (int*)take(BH_ * P_ * 4);
  int* gidx  = (int*)take(BH_ * 256 * 4);
  _Float16* kg  = (_Float16*)take((size_t)BH_ * 256 * HC_ * 2);
  _Float16* vgT = (_Float16*)take((size_t)BH_ * HC_ * 256 * 2);
  // f16 weight copies (converted once per call, ~1.1 MB total)
  _Float16* wq16  = (_Float16*)take((size_t)HID_ * C_ * 2);
  _Float16* wk16  = (_Float16*)take((size_t)HID_ * C_ * 2);
  _Float16* wv16  = (_Float16*)take((size_t)HID_ * C_ * 2);
  _Float16* wo16  = (_Float16*)take((size_t)C_ * HID_ * 2);
  _Float16* mlp16 = (_Float16*)take((size_t)C_ * 2 * C_ * 2);
  _Float16* c1w16 = (_Float16*)take((size_t)C_ * C_ * 2);
  _Float16* c2w16 = (_Float16*)take((size_t)C_ * C_ * 2);

  _Float16* q16   = (_Float16*)R1;
  _Float16* k16   = (_Float16*)R2;
  _Float16* att16 = (_Float16*)R3;
  _Float16* cat16 = (_Float16*)R4;
  float*    f_t   = (float*)R3;    // reuses att16 after wo
  _Float16* t16   = (_Float16*)R4; // reuses cat16 after mlp
  float*    buf0  = (float*)R1;    // reuses q16 after attention
  float*    buf1  = (float*)R2;    // reuses k16 after gather

  dim3 w32(32);
  // 0. convert weights to f16 once (keeps converts out of GEMM hot loops)
  cvt_f16_kernel<<<(HID_ * C_ + 255) / 256, 256, 0, stream>>>(wq, wq16, HID_ * C_);
  cvt_f16_kernel<<<(HID_ * C_ + 255) / 256, 256, 0, stream>>>(wk, wk16, HID_ * C_);
  cvt_f16_kernel<<<(HID_ * C_ + 255) / 256, 256, 0, stream>>>(wv, wv16, HID_ * C_);
  cvt_f16_kernel<<<(C_ * HID_ + 255) / 256, 256, 0, stream>>>(wo, wo16, C_ * HID_);
  cvt_f16_kernel<<<(C_ * 2 * C_ + 255) / 256, 256, 0, stream>>>(mlpw, mlp16, C_ * 2 * C_);
  cvt_f16_kernel<<<(C_ * C_ + 255) / 256, 256, 0, stream>>>(c1w, c1w16, C_ * C_);
  cvt_f16_kernel<<<(C_ * C_ + 255) / 256, 256, 0, stream>>>(c2w, c2w16, C_ * C_);

  // 1. channel LayerNorm -> (n,c) f16
  ln1_kernel<<<NTOT / 256, 256, 0, stream>>>(x, ln1g, ln1b, xn_t);
  // 2/3. q,k projections (WMMA), head-split layout; 16x64 strip per wave
  gemm16<MODE_HEADSPLIT><<<dim3(NTOT / 64, HID_ / 16), w32, 0, stream>>>(
      wq16, xn_t, nullptr, nullptr, nullptr, q16, C_);
  gemm16<MODE_HEADSPLIT><<<dim3(NTOT / 64, HID_ / 16), w32, 0, stream>>>(
      wk16, xn_t, nullptr, nullptr, nullptr, k16, C_);
  // 4/5. L2 normalize q, k
  l2norm_kernel<<<(BH_ * HW_) / 256, 256, 0, stream>>>(q16);
  l2norm_kernel<<<(BH_ * HW_) / 256, 256, 0, stream>>>(k16);
  // 6. qs = sum_q
  qs_kernel<<<BH_, 256, 0, stream>>>(q16, qs);
  // 7. row/col scores
  scores_kernel<<<dim3(128, BH_, 2), 64, 0, stream>>>(k16, qs, hs, wsS);
  // 8. top-16 rows & cols
  topk_kernel<<<BH_, 32, 0, stream>>>(hs, wsS, hidx, widx);
  // 9. gather k, build pixel index list
  gather_kernel<<<BH_, 256, 0, stream>>>(k16, hidx, widx, kg, gidx);
  // 10. v projection only at gathered pixels (WMMA) -> vgT (bh, 64, 256)
  gemm_vg<<<dim3(16, 4, BH_), w32, 0, stream>>>(wv16, xn_t, gidx, vgT);
  // 11. sparse attention (WMMA x64 per wave)
  attn_kernel<<<dim3(HW_ / 16, BH_), w32, 0, stream>>>(q16, kg, vgT, att16);
  // 12. wo projection -> cat[:,192:384]
  gemm16<MODE_CAT><<<dim3(NTOT / 64, C_ / 16), w32, 0, stream>>>(
      wo16, att16, bo, nullptr, nullptr, cat16, HID_);
  // 13. local depthwise conv -> cat[:,0:192]
  dwconv_kernel<1><<<(B_ * C_ * HW_) / 256, 256, 0, stream>>>(x, dww, dwb, cat16, nullptr);
  // 14. mlp 1x1 + bias + residual x -> f_t (n,c) f32
  gemm16<MODE_MLPF><<<dim3(NTOT / 64, C_ / 16), w32, 0, stream>>>(
      mlp16, cat16, mlpb, x, f_t, nullptr, 2 * C_);
  // 15. LayerNorm2 -> t16
  ln2_kernel<<<NTOT / 256, 256, 0, stream>>>(f_t, ln2g, ln2b, t16);
  // 16. c1 1x1 -> buf0 (NCHW)
  gemm16<MODE_NCHW><<<dim3(NTOT / 64, C_ / 16), w32, 0, stream>>>(
      c1w16, t16, c1b, nullptr, buf0, nullptr, C_);
  // 17. gelu(inorm) in place
  inorm_kernel<0><<<B_ * C_, 256, 0, stream>>>(buf0, nullptr, nullptr, nullptr);
  // 18. residual-block depthwise conv -> buf1
  dwconv_kernel<0><<<(B_ * C_ * HW_) / 256, 256, 0, stream>>>(buf0, resw, resb, nullptr, buf1);
  // 19. gelu(inorm) + residual -> t16
  inorm_kernel<1><<<B_ * C_, 256, 0, stream>>>(buf1, buf0, t16, nullptr);
  // 20. c2 1x1 -> buf0 (NCHW)
  gemm16<MODE_NCHW><<<dim3(NTOT / 64, C_ / 16), w32, 0, stream>>>(
      c2w16, t16, c2b, nullptr, buf0, nullptr, C_);
  // 21. final InstanceNorm -> out
  inorm_kernel<2><<<B_ * C_, 256, 0, stream>>>(buf0, nullptr, nullptr, out);
}
